// Fast_AutoEncoder_82463372083729
// MI455X (gfx1250) — matrix operand order repaired and checked by reading, not compiled, not run on hardware
//
#include <hip/hip_runtime.h>

// ---------------------------------------------------------------------------
// CDNA5 (gfx1250, wave32) hybrid AutoEncoder + LightGCN pipeline.
//  * GEMMs: v_wmma_f32_16x16x32_bf16 with split-bf16 (hi+lo) operands
//    pre-packed into separate planes (conversion out of the hot loop).
//  * Tile staging: async global->LDS (ASYNCcnt) double-buffered.
//  * Graph propagation: edge-parallel scatter with f32 atomics (state is
//    L2-resident: 2 x 82 MB < 192 MB L2).
// ---------------------------------------------------------------------------

typedef __attribute__((ext_vector_type(16))) __bf16 v16bf;
typedef __attribute__((ext_vector_type(8)))  float  v8f;

// GCC-style vector type matching the async-LDS builtin's parameter type.
typedef int v4i_gcc __attribute__((vector_size(16)));
typedef __attribute__((address_space(1))) v4i_gcc* gmem_v4i;
typedef __attribute__((address_space(3))) v4i_gcc* lds_v4i;

#if __has_builtin(__builtin_amdgcn_global_load_async_to_lds_b128)
#define GEMM_ASYNC 1
#else
#define GEMM_ASYNC 0
#endif

__device__ __forceinline__ void wait_async_all() {
#if GEMM_ASYNC
#if __has_builtin(__builtin_amdgcn_s_wait_asynccnt)
    __builtin_amdgcn_s_wait_asynccnt(0);
#else
    asm volatile("s_wait_asynccnt 0" ::: "memory");
#endif
#endif
}

// Copy 16 bytes (8 bf16) global -> LDS.
__device__ __forceinline__ void copy16_g2l(__bf16* l, const __bf16* g) {
#if GEMM_ASYNC
    v4i_gcc* gg = (v4i_gcc*)const_cast<__bf16*>(g);
    v4i_gcc* ll = (v4i_gcc*)l;
    __builtin_amdgcn_global_load_async_to_lds_b128(
        (gmem_v4i)gg, (lds_v4i)ll, /*offset=*/0, /*cpol=*/0);
#else
    *reinterpret_cast<uint4*>(l) = *reinterpret_cast<const uint4*>(g);
#endif
}

__device__ __forceinline__ void zero16_l(__bf16* l) {
    *reinterpret_cast<uint4*>(l) = make_uint4(0u, 0u, 0u, 0u);
}

__device__ __forceinline__ float sigmoidf_(float x) {
    return 1.0f / (1.0f + __expf(-x));
}

__device__ __forceinline__ void split_bf16(float x, __bf16& hi, __bf16& lo) {
    hi = (__bf16)x;
    lo = (__bf16)(x - (float)hi);
}

// Assemble a 16-element bf16 fragment from two 16B-aligned runs of 8 bf16.
__device__ __forceinline__ v16bf load_frag16(const __bf16* p0, const __bf16* p1) {
    union { uint4 q[2]; v16bf v; } u;
    u.q[0] = *reinterpret_cast<const uint4*>(p0);
    u.q[1] = *reinterpret_cast<const uint4*>(p1);
    return u.v;
}

// ---------------------------------------------------------------------------
// Encoder SpMM + sigmoid: one block per CSR row, 256 threads = one feature.
// ---------------------------------------------------------------------------
__global__ void enc_spmm_kernel(const int* __restrict__ Ap,
                                const int* __restrict__ Aj,
                                const float* __restrict__ Ax,
                                const float* __restrict__ ew0,
                                const float* __restrict__ eb0,
                                float* __restrict__ x0) {
    const int b = blockIdx.x;
    const int d = threadIdx.x;
    const int kBeg = Ap[b], kEnd = Ap[b + 1];
    float acc = 0.0f;
    for (int k = kBeg; k < kEnd; ++k) {
        const int   j = Aj[k];
        const float a = Ax[k];
        acc += a * ew0[(size_t)j * 256 + d];
    }
    x0[(size_t)b * 256 + d] = sigmoidf_(acc + eb0[d]);
}

// ---------------------------------------------------------------------------
// Pack f32 -> split bf16 planes (hi, lo).
// ---------------------------------------------------------------------------
__global__ void pack_bf16_kernel(const float* __restrict__ in,
                                 __bf16* __restrict__ hi,
                                 __bf16* __restrict__ lo,
                                 long long n) {
    const long long i = (long long)blockIdx.x * blockDim.x + threadIdx.x;
    if (i >= n) return;
    split_bf16(in[i], hi[i], lo[i]);
}

// ---------------------------------------------------------------------------
// 256x256 transpose + pack:  W[k][n] f32  ->  Wt_hi/lo[n][k] bf16.
// ---------------------------------------------------------------------------
__global__ void transpose_pack_kernel(const float* __restrict__ in,
                                      __bf16* __restrict__ outh,
                                      __bf16* __restrict__ outl) {
    const int k = blockIdx.x;            // 0..255
    const int n = threadIdx.x;           // 0..255
    const float v = in[(size_t)k * 256 + n];
    split_bf16(v, outh[(size_t)n * 256 + k], outl[(size_t)n * 256 + k]);
}

// ---------------------------------------------------------------------------
// Split-bf16 WMMA GEMM:  C[m][n] = act( sum_k A[m][k]*Bt[n][k] + bias[n] )
// Operands are pre-packed bf16 planes (rows x K, row-major).
// Block tile 128x128, K-step 32, 8 waves (2 M x 4 N), wave tile 64x32 =
// 4x2 WMMA tiles; 3 WMMAs per tile-step (ah*bh + ah*bl + al*bh).
// LDS tiles are double-buffered, filled via async global->LDS (ASYNCcnt).
// ---------------------------------------------------------------------------
__global__ __launch_bounds__(256)
void gemm_bf16split_kernel(const __bf16* __restrict__ Ah,
                           const __bf16* __restrict__ Al,
                           const __bf16* __restrict__ Bh,
                           const __bf16* __restrict__ Bl,
                           const float* __restrict__ bias,
                           float* __restrict__ C,
                           int M, int N, int K, int apply_sigmoid) {
    __shared__ __align__(16) __bf16 sAh[2][128 * 32];
    __shared__ __align__(16) __bf16 sAl[2][128 * 32];
    __shared__ __align__(16) __bf16 sBh[2][128 * 32];   // [n][k]
    __shared__ __align__(16) __bf16 sBl[2][128 * 32];

    const int t     = threadIdx.x;
    const int wid   = t >> 5;
    const int lane  = t & 31;
    const int waveM = wid >> 2;          // 0..1
    const int waveN = wid & 3;           // 0..3
    const int h     = lane >> 4;         // lane half
    const int l15   = lane & 15;

    const int mBase = blockIdx.y * 128;
    const int nBase = blockIdx.x * 128;

    v8f acc[4][2];
    const v8f vzero = {0.f, 0.f, 0.f, 0.f, 0.f, 0.f, 0.f, 0.f};
#pragma unroll
    for (int i = 0; i < 4; ++i)
#pragma unroll
        for (int j = 0; j < 2; ++j) acc[i][j] = vzero;

    // Stage one 128x32 K-slab of all four planes into LDS buffer `buf`.
    auto stage = [&](int buf, int k0) {
#pragma unroll
        for (int q = 0; q < 2; ++q) {
            const int id  = t + q * 256;     // 0..511 16B-granules per plane
            const int row = id >> 2;         // 0..127
            const int seg = id & 3;          // 4 x 8 bf16 per row
            const int o   = row * 32 + seg * 8;
            const int gm  = mBase + row;
            if (gm < M) {
                const size_t go = (size_t)gm * K + k0 + seg * 8;
                copy16_g2l(&sAh[buf][o], Ah + go);
                copy16_g2l(&sAl[buf][o], Al + go);
            } else {
                zero16_l(&sAh[buf][o]);
                zero16_l(&sAl[buf][o]);
            }
            const int gn = nBase + row;
            if (gn < N) {
                const size_t go = (size_t)gn * K + k0 + seg * 8;
                copy16_g2l(&sBh[buf][o], Bh + go);
                copy16_g2l(&sBl[buf][o], Bl + go);
            } else {
                zero16_l(&sBh[buf][o]);
                zero16_l(&sBl[buf][o]);
            }
        }
    };

    stage(0, 0);
    const int nsteps = K >> 5;
    for (int s = 0; s < nsteps; ++s) {
        wait_async_all();        // own async copies (into buf s&1) done
        __syncthreads();         // -> everyone's copies done, prev buf free
        const int buf = s & 1;
        if (s + 1 < nsteps) stage(buf ^ 1, (s + 1) << 5);   // prefetch next

        // B fragments: n = l15 (+16j), k = 16h..16h+15 (contiguous)
        v16bf bh[2], bl[2];
#pragma unroll
        for (int j = 0; j < 2; ++j) {
            const int n = waveN * 32 + j * 16 + l15;
            const __bf16* ph = &sBh[buf][n * 32 + h * 16];
            const __bf16* pl = &sBl[buf][n * 32 + h * 16];
            bh[j] = load_frag16(ph, ph + 8);
            bl[j] = load_frag16(pl, pl + 8);
        }
        // A fragments (k = 16h..16h+15, contiguous) + WMMA
#pragma unroll
        for (int i = 0; i < 4; ++i) {
            const int m = waveM * 64 + i * 16 + l15;
            const __bf16* pah = &sAh[buf][m * 32 + h * 16];
            const __bf16* pal = &sAl[buf][m * 32 + h * 16];
            const v16bf ah = load_frag16(pah, pah + 8);
            const v16bf al = load_frag16(pal, pal + 8);
#pragma unroll
            for (int j = 0; j < 2; ++j) {
                acc[i][j] = __builtin_amdgcn_wmma_f32_16x16x32_bf16(
                    false, ah, false, bh[j], (short)0, acc[i][j], false, false);
                acc[i][j] = __builtin_amdgcn_wmma_f32_16x16x32_bf16(
                    false, ah, false, bl[j], (short)0, acc[i][j], false, false);
                acc[i][j] = __builtin_amdgcn_wmma_f32_16x16x32_bf16(
                    false, al, false, bh[j], (short)0, acc[i][j], false, false);
            }
        }
    }

    // Epilogue: C/D layout -> row m = r + 8h, col n = l15.
#pragma unroll
    for (int i = 0; i < 4; ++i) {
#pragma unroll
        for (int j = 0; j < 2; ++j) {
            const int gn = nBase + waveN * 32 + j * 16 + l15;
            if (gn >= N) continue;
            const float bn = bias[gn];
#pragma unroll
            for (int r = 0; r < 8; ++r) {
                const int gm = mBase + waveM * 64 + i * 16 + r + 8 * h;
                if (gm >= M) continue;
                float v = acc[i][j][r] + bn;
                if (apply_sigmoid) v = sigmoidf_(v);
                C[(size_t)gm * N + gn] = v;
            }
        }
    }
}

// ---------------------------------------------------------------------------
// Edge-parallel SpMM (scatter): out[g_row[e]] += g_val[e] * in[g_col[e]].
// One wave per edge; each lane moves 8 of 256 features.
// ---------------------------------------------------------------------------
__global__ __launch_bounds__(256)
void spmm_edges_kernel(const int* __restrict__ g_row,
                       const int* __restrict__ g_col,
                       const float* __restrict__ g_val,
                       const float* __restrict__ inA,
                       const float* __restrict__ inB,
                       int split,
                       float* __restrict__ out,
                       int nnz) {
    const int e = blockIdx.x * (blockDim.x >> 5) + (threadIdx.x >> 5);
    if (e >= nnz) return;
    const int lane = threadIdx.x & 31;
    const int r = g_row[e];
    const int c = g_col[e];
    const float v = g_val[e];
    const float* src = (c < split) ? (inA + (size_t)c * 256)
                                   : (inB + (size_t)(c - split) * 256);
    float* dst = out + (size_t)r * 256;
    const int d = lane * 8;
    const float4 a = *reinterpret_cast<const float4*>(src + d);
    const float4 b = *reinterpret_cast<const float4*>(src + d + 4);
    atomicAdd(dst + d + 0, v * a.x);
    atomicAdd(dst + d + 1, v * a.y);
    atomicAdd(dst + d + 2, v * a.z);
    atomicAdd(dst + d + 3, v * a.w);
    atomicAdd(dst + d + 4, v * b.x);
    atomicAdd(dst + d + 5, v * b.y);
    atomicAdd(dst + d + 6, v * b.z);
    atomicAdd(dst + d + 7, v * b.w);
}

// ---------------------------------------------------------------------------
// light = (all_emb + cur1 + cur2)/3, written in place into cur1.
// ---------------------------------------------------------------------------
__global__ void light_kernel(const float* __restrict__ user_emb,
                             const float* __restrict__ item_emb,
                             float* __restrict__ cur1,
                             const float* __restrict__ cur2,
                             int U) {
    const int r = blockIdx.x;
    const int d = threadIdx.x;
    const size_t i = (size_t)r * 256 + d;
    const float emb = (r < U) ? user_emb[i]
                              : item_emb[(size_t)(r - U) * 256 + d];
    cur1[i] = (emb + cur1[i] + cur2[i]) * (1.0f / 3.0f);
}

// ---------------------------------------------------------------------------
// xi = 0.5*items_avg[idx-B : idx-B+B] + 0.5*x2  (P = 0.5)
// ---------------------------------------------------------------------------
__global__ void xi_kernel(const float* __restrict__ light,   // cur1
                          const float* __restrict__ x2,
                          const int* __restrict__ idx_p,
                          float* __restrict__ xi,
                          int U, int Bn) {
    const int m = blockIdx.x;
    const int d = threadIdx.x;
    const int base = idx_p[0] - Bn;
    const size_t li = (size_t)(U + base + m) * 256 + d;
    const size_t mi = (size_t)m * 256 + d;
    xi[mi] = 0.5f * light[li] + 0.5f * x2[mi];
}

// ---------------------------------------------------------------------------
// Orchestration (graph-capture safe: kernel launches + hipMemsetAsync only).
// ---------------------------------------------------------------------------
extern "C" void kernel_launch(void* const* d_in, const int* in_sizes, int n_in,
                              void* d_out, int out_size, void* d_ws, size_t ws_size,
                              hipStream_t stream) {
    const int*   Ap       = (const int*)  d_in[0];
    const int*   Aj       = (const int*)  d_in[1];
    const float* Ax       = (const float*)d_in[2];
    const int*   g_row    = (const int*)  d_in[3];
    const int*   g_col    = (const int*)  d_in[4];
    const float* g_val    = (const float*)d_in[5];
    const float* ew0      = (const float*)d_in[6];
    const float* eb0      = (const float*)d_in[7];
    const float* ew1      = (const float*)d_in[8];
    const float* eb1      = (const float*)d_in[9];
    const float* dw0      = (const float*)d_in[10];
    const float* db0      = (const float*)d_in[11];
    const float* db1      = (const float*)d_in[12];
    const float* user_emb = (const float*)d_in[13];
    const float* item_emb = (const float*)d_in[14];
    const int*   idx_p    = (const int*)  d_in[15];

    const int Bn   = in_sizes[0] - 1;          // 2048
    const int nnzG = in_sizes[3];              // 800000
    const int D    = in_sizes[7];              // 256
    const int U    = in_sizes[12];             // 50000
    const int I    = in_sizes[14] / D;         // 30000
    const int V    = U + I;                    // 80000

    const size_t nBD = (size_t)Bn * D;
    const size_t nDD = (size_t)D * D;
    const size_t nVD = (size_t)V * D;
    const size_t nUD = (size_t)U * D;

    // f32 region
    float* ws   = (float*)d_ws;
    float* x0   = ws;                          // Bn*D
    float* x1   = x0 + nBD;
    float* x2   = x1 + nBD;
    float* xi   = x2 + nBD;
    float* cur1 = xi + nBD;                    // V*D
    float* cur2 = cur1 + nVD;                  // V*D
    // bf16 region (starts 16B-aligned: preceding float count is a mult. of 4)
    __bf16* bp   = (__bf16*)(cur2 + nVD);
    __bf16* x0h  = bp;            __bf16* x0l  = x0h + nBD;
    __bf16* x1h  = x0l + nBD;     __bf16* x1l  = x1h + nBD;
    __bf16* xih  = x1l + nBD;     __bf16* xil  = xih + nBD;
    __bf16* bt1h = xil + nBD;     __bf16* bt1l = bt1h + nDD;
    __bf16* bt2h = bt1l + nDD;    __bf16* bt2l = bt2h + nDD;
    __bf16* wuh  = bt2l + nDD;    __bf16* wul  = wuh + nUD;
    float* out  = (float*)d_out;

    // 1) encoder SpMM + sigmoid, then pack
    enc_spmm_kernel<<<Bn, 256, 0, stream>>>(Ap, Aj, Ax, ew0, eb0, x0);
    pack_bf16_kernel<<<(int)(nBD / 256), 256, 0, stream>>>(x0, x0h, x0l,
                                                           (long long)nBD);

    // 2) small dense layers (transpose+pack weights once)
    transpose_pack_kernel<<<D, D, 0, stream>>>(ew1, bt1h, bt1l);
    transpose_pack_kernel<<<D, D, 0, stream>>>(dw0, bt2h, bt2l);
    {
        dim3 grid((D + 127) / 128, (Bn + 127) / 128);
        gemm_bf16split_kernel<<<grid, 256, 0, stream>>>(
            x0h, x0l, bt1h, bt1l, eb1, x1, Bn, D, D, 1);
        pack_bf16_kernel<<<(int)(nBD / 256), 256, 0, stream>>>(
            x1, x1h, x1l, (long long)nBD);
        gemm_bf16split_kernel<<<grid, 256, 0, stream>>>(
            x1h, x1l, bt2h, bt2l, db0, x2, Bn, D, D, 1);
    }

    // 3) LightGCN propagation: two scatter-SpMM passes (L2-resident atomics)
    const size_t embBytes = nVD * sizeof(float);
    (void)hipMemsetAsync(cur1, 0, embBytes, stream);
    {
        const int blocks = (nnzG + 7) / 8;
        spmm_edges_kernel<<<blocks, 256, 0, stream>>>(g_row, g_col, g_val,
                                                      user_emb, item_emb, U,
                                                      cur1, nnzG);
        (void)hipMemsetAsync(cur2, 0, embBytes, stream);
        spmm_edges_kernel<<<blocks, 256, 0, stream>>>(g_row, g_col, g_val,
                                                      cur1, cur1, 0,
                                                      cur2, nnzG);
    }
    light_kernel<<<V, 256, 0, stream>>>(user_emb, item_emb, cur1, cur2, U);

    // 4) xi = 0.5*items_avg + 0.5*x2 ; pack GEMM operands
    xi_kernel<<<Bn, 256, 0, stream>>>(cur1, x2, idx_p, xi, U, Bn);
    pack_bf16_kernel<<<(int)(nBD / 256), 256, 0, stream>>>(xi, xih, xil,
                                                           (long long)nBD);
    pack_bf16_kernel<<<(int)((nUD + 255) / 256), 256, 0, stream>>>(
        cur1, wuh, wul, (long long)nUD);     // w_users = light[:U], [n][k]

    // 5) big GEMM: out = xi @ w_users^T + db1
    {
        dim3 grid((U + 127) / 128, (Bn + 127) / 128);
        gemm_bf16split_kernel<<<grid, 256, 0, stream>>>(
            xih, xil, wuh, wul, db1, out, Bn, U, D, 0);
    }
}
